// DNM_Linear_M3_9689446220607
// MI455X (gfx1250) — matrix operand order repaired and checked by reading, not compiled, not run on hardware
//
#include <hip/hip_runtime.h>

typedef __attribute__((ext_vector_type(16))) _Float16 v16h;
typedef __attribute__((ext_vector_type(8)))  float    v8f;
typedef __attribute__((ext_vector_type(2)))  float    f32x2;

#define B_    64
#define OUT_  512
#define M_    5
#define IN_   1024

#define LOG2E 1.4426950408889634f

// CDNA5 v_tanh_f32; fallback to exp2/rcp identity if builtin missing.
__device__ __forceinline__ float fast_tanh(float t) {
#if __has_builtin(__builtin_amdgcn_tanhf)
    return __builtin_amdgcn_tanhf(t);
#else
    // tanh(t) = 1 - 2/(1 + e^{2t})
    float u = __builtin_amdgcn_exp2f(t * (2.0f * LOG2E));
    return __builtin_fmaf(-2.0f, __builtin_amdgcn_rcpf(1.0f + u), 1.0f);
#endif
}

__global__ __launch_bounds__(256)
void dnm_linear_m3_wmma(const float* __restrict__ x,
                        const float* __restrict__ W,
                        const float* __restrict__ q,
                        const float* __restrict__ W2,
                        const float* __restrict__ kk,
                        const float* __restrict__ qs,
                        float* __restrict__ out)
{
    const int tid  = threadIdx.x;
    const int lane = tid & 31;
    const int wave = tid >> 5;

    // 2048 (o, b-tile) pairs; 8 waves/block, waves in a block share `o` pairs
    const int gw = blockIdx.x * 8 + wave;     // 0..2047
    const int o  = gw >> 2;                   // 0..511
    const int b0 = (gw & 3) << 4;             // 0,16,32,48

    const int half_hi = lane >> 4;            // 0: lanes 0-15, 1: lanes 16-31
    const int arow    = b0 + (lane & 15);     // A-matrix row (batch index)
    const int koffA   = half_hi * 8;          // A-matrix K-offset per ISA layout
    const int koffB   = half_hi * 8;          // B-matrix K-offset per ISA layout

    const float* __restrict__ xrow  = x + arow * IN_;
    const float* __restrict__ wbase = W + o * (M_ * IN_);
    const float* __restrict__ qbase = q + o * (M_ * IN_);

    // ---- S = sum(W2) : wave-invariant scalar, computed once ----
    float s_part = 0.0f;
    {
        const float4* __restrict__ w2v4 = (const float4*)W2;
#pragma unroll
        for (int j = 0; j < 8; ++j) {
            float4 t = w2v4[lane * 8 + j];
            s_part += (t.x + t.y) + (t.z + t.w);
        }
#pragma unroll
        for (int off = 16; off > 0; off >>= 1)
            s_part += __shfl_xor(s_part, off, 32);
    }
    const float S4 = 0.25f * s_part;          // 0.25 * sum(W2)

    v8f acc[M_];
#pragma unroll
    for (int m = 0; m < M_; ++m) acc[m] = (v8f){0.f,0.f,0.f,0.f,0.f,0.f,0.f,0.f};

    for (int i0 = 0; i0 < IN_; i0 += 32) {
        // ---- B matrix: 32x16 f16, every column = the W2 chunk ----
        // lanes 0-15: K = {0..7, 16..23}; lanes 16-31: +8
        const float4 w2a = *(const float4*)(W2 + i0 + koffB);
        const float4 w2b = *(const float4*)(W2 + i0 + koffB + 4);
        const float4 w2c = *(const float4*)(W2 + i0 + 16 + koffB);
        const float4 w2d = *(const float4*)(W2 + i0 + 16 + koffB + 4);
        float w2g[16] = { w2a.x, w2a.y, w2a.z, w2a.w,
                          w2b.x, w2b.y, w2b.z, w2b.w,
                          w2c.x, w2c.y, w2c.z, w2c.w,
                          w2d.x, w2d.y, w2d.z, w2d.w };
        v16h bm;
#pragma unroll
        for (int e = 0; e < 16; ++e) bm[e] = (_Float16)w2g[e];

        // ---- x chunk for this lane's K slots (reused for all 5 branches) ----
        // lanes 0-15: K = {i0+0..7, i0+16..23}; lanes 16-31: +8
        // Pre-scaled by 0.25 once per chunk (amortized over the 5 branches),
        // kept as f32x2 pairs to encourage v_pk_* packed math.
        const float4 x0a = *(const float4*)(xrow + i0 + koffA);
        const float4 x0b = *(const float4*)(xrow + i0 + koffA + 4);
        const float4 x1a = *(const float4*)(xrow + i0 + 16 + koffA);
        const float4 x1b = *(const float4*)(xrow + i0 + 16 + koffA + 4);
        f32x2 xg[8] = { (f32x2){x0a.x, x0a.y}, (f32x2){x0a.z, x0a.w},
                        (f32x2){x0b.x, x0b.y}, (f32x2){x0b.z, x0b.w},
                        (f32x2){x1a.x, x1a.y}, (f32x2){x1a.z, x1a.w},
                        (f32x2){x1b.x, x1b.y}, (f32x2){x1b.z, x1b.w} };
#pragma unroll
        for (int p = 0; p < 8; ++p) xg[p] = xg[p] * 0.25f;  // v_pk_mul_f32

#pragma unroll
        for (int m = 0; m < M_; ++m) {
            const float* __restrict__ wrow = wbase + m * IN_;
            const float* __restrict__ qrow = qbase + m * IN_;

            const float4 w0a = *(const float4*)(wrow + i0 + koffA);
            const float4 w0b = *(const float4*)(wrow + i0 + koffA + 4);
            const float4 w1a = *(const float4*)(wrow + i0 + 16 + koffA);
            const float4 w1b = *(const float4*)(wrow + i0 + 16 + koffA + 4);
            const float4 q0a = *(const float4*)(qrow + i0 + koffA);
            const float4 q0b = *(const float4*)(qrow + i0 + koffA + 4);
            const float4 q1a = *(const float4*)(qrow + i0 + 16 + koffA);
            const float4 q1b = *(const float4*)(qrow + i0 + 16 + koffA + 4);

            f32x2 wg[8] = { (f32x2){w0a.x, w0a.y}, (f32x2){w0a.z, w0a.w},
                            (f32x2){w0b.x, w0b.y}, (f32x2){w0b.z, w0b.w},
                            (f32x2){w1a.x, w1a.y}, (f32x2){w1a.z, w1a.w},
                            (f32x2){w1b.x, w1b.y}, (f32x2){w1b.z, w1b.w} };
            f32x2 qg[8] = { (f32x2){q0a.x, q0a.y}, (f32x2){q0a.z, q0a.w},
                            (f32x2){q0b.x, q0b.y}, (f32x2){q0b.z, q0b.w},
                            (f32x2){q1a.x, q1a.y}, (f32x2){q1a.z, q1a.w},
                            (f32x2){q1b.x, q1b.y}, (f32x2){q1b.z, q1b.w} };

            // A holds raw tanh(0.25*(x*w - q)) in f16; the affine 0.5*t + 0.5
            // of the sigmoid is linear, so it is folded past the WMMA:
            //   sum_i sigmoid*W2 = 0.5*acc + 0.5*sum(W2)
            // t-pair = (0.25*x)*w - 0.25*q  -> v_pk_fma_f32 per pair
            v16h am;
#pragma unroll
            for (int p = 0; p < 8; ++p) {
                f32x2 t = xg[p] * wg[p] - qg[p] * 0.25f;
                am[2 * p]     = (_Float16)fast_tanh(t.x);
                am[2 * p + 1] = (_Float16)fast_tanh(t.y);
            }

            acc[m] = __builtin_amdgcn_wmma_f32_16x16x32_f16(
                         false, am, false, bm, (short)0, acc[m], false, false);
        }
    }

    // Each lane holds dendrite partials for rows (half_hi*8 + v), v=0..7:
    //   d_m = 0.5*acc_m + 0.5*S
    //   sigmoid(d_m) = 0.5 + 0.5*tanh(0.25*acc_m + 0.25*S)
    //   out = k*(sum_m sigmoid(d_m) - qs) = (0.5*k)*T + k*(M/2 - qs)
    const float ks = kk[0];
    const float q0 = qs[0];
    const float c1 = 0.5f * ks;
    const float c0 = ks * (0.5f * (float)M_ - q0);

    float ys[8];
#pragma unroll
    for (int v = 0; v < 8; ++v) {
        float T = 0.0f;
#pragma unroll
        for (int m = 0; m < M_; ++m) {
            float t = __builtin_fmaf(acc[m][v], 0.25f, S4);
            T += fast_tanh(t);
        }
        ys[v] = __builtin_fmaf(c1, T, c0);
    }

    if ((lane & 15) == 0) {
        const int rbase = b0 + half_hi * 8;
#pragma unroll
        for (int v = 0; v < 8; ++v) {
            out[(rbase + v) * OUT_ + o] = ys[v];
        }
    }
}

extern "C" void kernel_launch(void* const* d_in, const int* in_sizes, int n_in,
                              void* d_out, int out_size, void* d_ws, size_t ws_size,
                              hipStream_t stream) {
    const float* x  = (const float*)d_in[0];   // [B, IN]
    const float* W  = (const float*)d_in[1];   // [OUT, M, IN]
    const float* q  = (const float*)d_in[2];   // [OUT, M, IN]
    const float* W2 = (const float*)d_in[3];   // [IN]
    const float* kk = (const float*)d_in[4];   // [1]
    const float* qs = (const float*)d_in[5];   // [1]
    float* out = (float*)d_out;                // [B, OUT]

    // 2048 waves total = 256 blocks * 8 waves
    dnm_linear_m3_wmma<<<256, 256, 0, stream>>>(x, W, q, W2, kk, qs, out);
}
